// Point_Encoder_3496103378966
// MI455X (gfx1250) — compile-verified, matrix operand order checked by reading, not run on hardware
//
#include <hip/hip_runtime.h>
#include <hip/hip_bf16.h>
#include <math.h>

// ---------------------------------------------------------------------------
// PointNet++ MSG encoder for MI455X (gfx1250, wave32, WMMA bf16 matrix core)
// ---------------------------------------------------------------------------

typedef __attribute__((ext_vector_type(16))) __bf16 v16bf;
typedef __attribute__((ext_vector_type(8)))  __bf16 v8bf;
typedef __attribute__((ext_vector_type(8)))  float  v8f;

#define CEILDIV(a,b) (((a)+(b)-1)/(b))

static inline int cp32(int c){ return (c + 31) & ~31; }   // pad K-dim to 32
static inline int np16(int c){ return (c + 15) & ~15; }   // pad N-dim to 16

// ---------------------------------------------------------------------------
// 0) input transpose: xyz (B,6,N) -> cur_xyz (B,N,3), cur_feat (B,N,6);
//    also emit out0 = xyz[:, :3, :] and out1 = xyz (channel-first copies)
// ---------------------------------------------------------------------------
__global__ void k_prep(const float* __restrict__ xin, float* __restrict__ cxyz,
                       float* __restrict__ cfeat, float* __restrict__ out0,
                       float* __restrict__ out1, int B_, int N_) {
  int i = blockIdx.x * blockDim.x + threadIdx.x;   // over B*N
  if (i >= B_ * N_) return;
  int n = i % N_, b = i / N_;
  for (int c = 0; c < 6; ++c) {
    float v = xin[((size_t)b * 6 + c) * N_ + n];
    cfeat[(size_t)i * 6 + c] = v;
    out1[((size_t)b * 6 + c) * N_ + n] = v;
    if (c < 3) {
      cxyz[(size_t)i * 3 + c] = v;
      out0[((size_t)b * 3 + c) * N_ + n] = v;
    }
  }
}

// ---------------------------------------------------------------------------
// 1) farthest point sampling: one block per batch, dist array in LDS,
//    sequential npoint iterations, tie-break argmax -> lowest index.
// ---------------------------------------------------------------------------
__global__ void k_fps(const float* __restrict__ xyz, int N_, int np, int* __restrict__ fidx) {
  __shared__ float dist[2048];
  __shared__ float rv[256];
  __shared__ int   ri[256];
  int b = blockIdx.x;
  int tid = threadIdx.x;
  const float* X = xyz + (size_t)b * N_ * 3;
  for (int i = tid; i < N_; i += 256) dist[i] = 1e10f;
  __syncthreads();
  int far = 0;
  for (int s = 0; s < np; ++s) {
    if (tid == 0) fidx[b * np + s] = far;
    float cx = X[far * 3 + 0], cy = X[far * 3 + 1], cz = X[far * 3 + 2];
    float best = -1.0f; int bi = 0x7fffffff;
    for (int i = tid; i < N_; i += 256) {
      float dx = X[i * 3 + 0] - cx, dy = X[i * 3 + 1] - cy, dz = X[i * 3 + 2] - cz;
      float d = dx * dx + dy * dy + dz * dz;
      float nd = fminf(dist[i], d);
      dist[i] = nd;
      if (nd > best) { best = nd; bi = i; }
    }
    rv[tid] = best; ri[tid] = bi;
    __syncthreads();
    for (int off = 128; off > 0; off >>= 1) {
      if (tid < off) {
        float ov = rv[tid + off]; int oi = ri[tid + off];
        if (ov > rv[tid] || (ov == rv[tid] && oi < ri[tid])) { rv[tid] = ov; ri[tid] = oi; }
      }
      __syncthreads();
    }
    far = ri[0];
    __syncthreads();
  }
}

// ---------------------------------------------------------------------------
// 2) gather new_xyz (B,S,3) and write transposed (B,3,S) to output
// ---------------------------------------------------------------------------
__global__ void k_newxyz(const float* __restrict__ xyz, const int* __restrict__ fidx,
                         float* __restrict__ nxyz, float* __restrict__ outp,
                         int B_, int Nin, int S) {
  int i = blockIdx.x * blockDim.x + threadIdx.x;   // B*S
  if (i >= B_ * S) return;
  int b = i / S, s = i % S;
  int j = fidx[i];
  for (int c = 0; c < 3; ++c) {
    float v = xyz[((size_t)b * Nin + j) * 3 + c];
    nxyz[(size_t)i * 3 + c] = v;
    outp[((size_t)b * 3 + c) * S + s] = v;
  }
}

// ---------------------------------------------------------------------------
// 3) ball query: thread per center, ascending scan == sort-by-index + take K.
//    Empty ball -> fill with Nin-1 (JAX gather clamp of index==N).
// ---------------------------------------------------------------------------
__global__ void k_ball(const float* __restrict__ xyz, const float* __restrict__ nxyz,
                       int* __restrict__ bidx, int B_, int Nin, int S, int K, float r2) {
  int i = blockIdx.x * blockDim.x + threadIdx.x;   // B*S
  if (i >= B_ * S) return;
  int b = i / S;
  const float* X = xyz + (size_t)b * Nin * 3;
  float cx = nxyz[(size_t)i * 3 + 0], cy = nxyz[(size_t)i * 3 + 1], cz = nxyz[(size_t)i * 3 + 2];
  int* out = bidx + (size_t)i * K;
  int cnt = 0, first = Nin - 1;
  for (int j = 0; j < Nin && cnt < K; ++j) {
    float dx = X[j * 3 + 0] - cx, dy = X[j * 3 + 1] - cy, dz = X[j * 3 + 2] - cz;
    if (dx * dx + dy * dy + dz * dz <= r2) { if (cnt == 0) first = j; out[cnt++] = j; }
  }
  for (; cnt < K; ++cnt) out[cnt] = first;
}

// ---------------------------------------------------------------------------
// 4) group + concat (features first, then relative xyz), cast to bf16,
//    zero-pad channels to CP. X layout: row-major [B*S*K][CP].
// ---------------------------------------------------------------------------
__global__ void k_group(const float* __restrict__ xyz, const float* __restrict__ points,
                        const float* __restrict__ nxyz, const int* __restrict__ bidx,
                        __bf16* __restrict__ X, int B_, int Nin, int Cp, int S, int K, int CP) {
  int row = blockIdx.x * blockDim.x + threadIdx.x;   // B*S*K
  if (row >= B_ * S * K) return;
  int s = (row / K) % S, b = row / (S * K);
  int j = bidx[row];
  if (j >= Nin) j = Nin - 1;
  const float* p = points + ((size_t)b * Nin + j) * Cp;
  const float* g = xyz + ((size_t)b * Nin + j) * 3;
  const float* nx = nxyz + ((size_t)b * S + s) * 3;
  __bf16* x = X + (size_t)row * CP;
  int c = 0;
  for (; c < Cp; ++c) x[c] = (__bf16)p[c];
  x[c++] = (__bf16)(g[0] - nx[0]);
  x[c++] = (__bf16)(g[1] - nx[1]);
  x[c++] = (__bf16)(g[2] - nx[2]);
  for (; c < CP; ++c) x[c] = (__bf16)0.0f;
}

// ---------------------------------------------------------------------------
// 5) weight cast fp32 (O,Cin) -> bf16 (NPad,CP), zero padded
// ---------------------------------------------------------------------------
__global__ void k_castw(const float* __restrict__ w, __bf16* __restrict__ wb,
                        int O, int Cin, int CP, int NPad) {
  int o = blockIdx.x * blockDim.x + threadIdx.x;
  if (o >= NPad) return;
  __bf16* d = wb + (size_t)o * CP;
  if (o < O) {
    const float* s = w + (size_t)o * Cin;
    for (int c = 0; c < CP; ++c) d[c] = (__bf16)((c < Cin) ? s[c] : 0.0f);
  } else {
    for (int c = 0; c < CP; ++c) d[c] = (__bf16)0.0f;
  }
}

// ---------------------------------------------------------------------------
// 6) WMMA GEMM, register-blocked over the full output-channel stripe:
//    each wave computes one 16-row M-tile x NT 16-col N-tiles, NT = NPad/16.
//    Accumulators: NT x 8 f32 VGPRs. A k-chunk loaded ONCE per kb (was NT
//    times with per-tile waves) -> up to 16x less HBM traffic on the big
//    activation buffers, which are the bandwidth bottleneck at 23.3 TB/s.
//    Weights (<=278 KB) stay L2-resident so their re-reads are cheap.
//    Operand lane layout per CDNA5 ISA 16-bit A 16x32 table:
//      lanes 0-15 : row = lane,    halves [0..7]=K 0..7,  [8..15]=K 16..23
//      lanes 16-31: row = lane-16, halves [0..7]=K 8..15, [8..15]=K 24..31
//    (B mirrored with N = lane&15). D: c[r] -> row tm*16+r+8*(lane>>4).
// ---------------------------------------------------------------------------
template <int NT>
__global__ void k_gemm_bf16_nt(const __bf16* __restrict__ X, const __bf16* __restrict__ W,
                               const float* __restrict__ bias, float* __restrict__ Y,
                               int CP, int NPad, int O) {
  int tm = blockIdx.x;
  int lane = threadIdx.x;
  int hl = lane >> 4;
  int l16 = lane & 15;
  const __bf16* xrow = X + (size_t)(tm * 16 + l16) * CP;
  const __bf16* wbase = W + (size_t)l16 * CP;
  v8f acc[NT];
  #pragma unroll
  for (int t = 0; t < NT; ++t) { v8f z = {}; acc[t] = z; }
  for (int kb = 0; kb < CP; kb += 32) {
    int c0 = kb + hl * 8;
    v8bf alo = *(const v8bf*)(xrow + c0);
    v8bf ahi = *(const v8bf*)(xrow + c0 + 16);
    v16bf a;
    #pragma unroll
    for (int i = 0; i < 8; ++i) { a[i] = alo[i]; a[i + 8] = ahi[i]; }
    #pragma unroll
    for (int t = 0; t < NT; ++t) {
      const __bf16* wrow = wbase + (size_t)t * 16 * CP;
      v8bf blo = *(const v8bf*)(wrow + c0);
      v8bf bhi = *(const v8bf*)(wrow + c0 + 16);
      v16bf bm;
      #pragma unroll
      for (int i = 0; i < 8; ++i) { bm[i] = blo[i]; bm[i + 8] = bhi[i]; }
      acc[t] = __builtin_amdgcn_wmma_f32_16x16x32_bf16(false, a, false, bm, (short)0,
                                                       acc[t], false, false);
    }
  }
  #pragma unroll
  for (int t = 0; t < NT; ++t) {
    int n = t * 16 + l16;
    float bv = (n < O) ? bias[n] : 0.0f;
    #pragma unroll
    for (int r = 0; r < 8; ++r) {
      int m = tm * 16 + r + hl * 8;
      Y[(size_t)m * NPad + n] = (n < O) ? (acc[t][r] + bv) : 0.0f;
    }
  }
}

// generic fallback (one tile per wave) for unexpected shapes
__global__ void k_gemm_bf16_1t(const __bf16* __restrict__ X, const __bf16* __restrict__ W,
                               const float* __restrict__ bias, float* __restrict__ Y,
                               int CP, int NPad, int O) {
  int tm = blockIdx.x, tn = blockIdx.y;
  int lane = threadIdx.x;
  int hl = lane >> 4;
  int l16 = lane & 15;
  const __bf16* xrow = X + (size_t)(tm * 16 + l16) * CP;
  const __bf16* wrow = W + (size_t)(tn * 16 + l16) * CP;
  v8f acc = {};
  for (int kb = 0; kb < CP; kb += 32) {
    int c0 = kb + hl * 8;
    v8bf alo = *(const v8bf*)(xrow + c0);
    v8bf ahi = *(const v8bf*)(xrow + c0 + 16);
    v8bf blo = *(const v8bf*)(wrow + c0);
    v8bf bhi = *(const v8bf*)(wrow + c0 + 16);
    v16bf a, bm;
    #pragma unroll
    for (int i = 0; i < 8; ++i) { a[i] = alo[i]; a[i + 8] = ahi[i]; bm[i] = blo[i]; bm[i + 8] = bhi[i]; }
    acc = __builtin_amdgcn_wmma_f32_16x16x32_bf16(false, a, false, bm, (short)0, acc, false, false);
  }
  int n = tn * 16 + l16;
  float bv = (n < O) ? bias[n] : 0.0f;
  #pragma unroll
  for (int r = 0; r < 8; ++r) {
    int m = tm * 16 + r + hl * 8;
    Y[(size_t)m * NPad + n] = (n < O) ? (acc[r] + bv) : 0.0f;
  }
}

static void launch_gemm(const __bf16* X, const __bf16* W, const float* bias, float* Y,
                        int M, int CP, int NPad, int O, hipStream_t stream) {
  int nt = NPad / 16;
  dim3 g1(M / 16);
  switch (nt) {
    case 2:  k_gemm_bf16_nt<2> <<<g1, 32, 0, stream>>>(X, W, bias, Y, CP, NPad, O); break;
    case 4:  k_gemm_bf16_nt<4> <<<g1, 32, 0, stream>>>(X, W, bias, Y, CP, NPad, O); break;
    case 6:  k_gemm_bf16_nt<6> <<<g1, 32, 0, stream>>>(X, W, bias, Y, CP, NPad, O); break;
    case 8:  k_gemm_bf16_nt<8> <<<g1, 32, 0, stream>>>(X, W, bias, Y, CP, NPad, O); break;
    case 13: k_gemm_bf16_nt<13><<<g1, 32, 0, stream>>>(X, W, bias, Y, CP, NPad, O); break;
    case 16: k_gemm_bf16_nt<16><<<g1, 32, 0, stream>>>(X, W, bias, Y, CP, NPad, O); break;
    default: {
      dim3 g2(M / 16, NPad / 16);
      k_gemm_bf16_1t<<<g2, 32, 0, stream>>>(X, W, bias, Y, CP, NPad, O);
    }
  }
}

// ---------------------------------------------------------------------------
// 7) per-channel batch stats: block partial sums + few atomics
// ---------------------------------------------------------------------------
__global__ void k_colstats(const float* __restrict__ Y, int M, int NPad,
                           float* __restrict__ sum, float* __restrict__ sumsq) {
  __shared__ float bs[256], bq[256];
  int ch = blockIdx.x;
  int start = blockIdx.y * 8192;
  int end = start + 8192; if (end > M) end = M;
  float s = 0.f, q = 0.f;
  for (int r = start + threadIdx.x; r < end; r += blockDim.x) {
    float v = Y[(size_t)r * NPad + ch];
    s += v; q += v * v;
  }
  bs[threadIdx.x] = s; bq[threadIdx.x] = q;
  __syncthreads();
  for (int off = 128; off > 0; off >>= 1) {
    if (threadIdx.x < off) { bs[threadIdx.x] += bs[threadIdx.x + off]; bq[threadIdx.x] += bq[threadIdx.x + off]; }
    __syncthreads();
  }
  if (threadIdx.x == 0) { atomicAdd(&sum[ch], bs[0]); atomicAdd(&sumsq[ch], bq[0]); }
}

__global__ void k_finstats(const float* __restrict__ sum, const float* __restrict__ sumsq,
                           int M, int O, float* __restrict__ mean, float* __restrict__ rstd) {
  int c = blockIdx.x * blockDim.x + threadIdx.x;
  if (c >= O) return;
  float m = sum[c] / (float)M;
  float v = sumsq[c] / (float)M - m * m;
  mean[c] = m;
  rstd[c] = rsqrtf(v + 1e-5f);
}

// ---------------------------------------------------------------------------
// 8) BN + ReLU + cast to bf16 for next layer's A operand (zero padded)
// ---------------------------------------------------------------------------
__global__ void k_bnrelu(const float* __restrict__ Y, const float* __restrict__ mean,
                         const float* __restrict__ rstd, const float* __restrict__ gamma,
                         const float* __restrict__ beta, __bf16* __restrict__ Xn,
                         int M, int NPad, int O, int CPo) {
  int row = blockIdx.x * blockDim.x + threadIdx.x;
  if (row >= M) return;
  const float* y = Y + (size_t)row * NPad;
  __bf16* x = Xn + (size_t)row * CPo;
  for (int c = 0; c < CPo; ++c) {
    float v = 0.f;
    if (c < O) {
      v = (y[c] - mean[c]) * rstd[c] * gamma[c] + beta[c];
      v = v > 0.f ? v : 0.f;
    }
    x[c] = (__bf16)v;
  }
}

// ---------------------------------------------------------------------------
// 9) last layer of a branch: BN + ReLU fused with max over K neighbors
// ---------------------------------------------------------------------------
__global__ void k_maxk(const float* __restrict__ Y, const float* __restrict__ mean,
                       const float* __restrict__ rstd, const float* __restrict__ gamma,
                       const float* __restrict__ beta, float* __restrict__ feat,
                       int BS, int K, int NPad, int O, int Ctot, int choff) {
  int i = blockIdx.x * blockDim.x + threadIdx.x;   // BS*O
  if (i >= BS * O) return;
  int bs = i / O, c = i % O;
  const float* y = Y + (size_t)bs * K * NPad + c;
  float me = mean[c], rs = rstd[c], g = gamma[c], be = beta[c];
  float mx = -1e30f;
  for (int k = 0; k < K; ++k) {
    float v = (y[(size_t)k * NPad] - me) * rs * g + be;
    v = v > 0.f ? v : 0.f;
    if (v > mx) mx = v;
  }
  feat[(size_t)bs * Ctot + choff + c] = mx;
}

// ---------------------------------------------------------------------------
// 10) feat (B,S,C) -> output (B,C,S)
// ---------------------------------------------------------------------------
__global__ void k_featout(const float* __restrict__ feat, float* __restrict__ outp,
                          int B_, int S, int C) {
  int i = blockIdx.x * blockDim.x + threadIdx.x;
  if (i >= B_ * S * C) return;
  int c = i % C, s = (i / C) % S, b = i / (S * C);
  outp[((size_t)b * C + c) * S + s] = feat[i];
}

// ---------------------------------------------------------------------------
// Host orchestration
// ---------------------------------------------------------------------------
extern "C" void kernel_launch(void* const* d_in, const int* in_sizes, int n_in,
                              void* d_out, int out_size, void* d_ws, size_t ws_size,
                              hipStream_t stream) {
  const int B = 8, N0 = 2048;
  const float* xin = (const float*)d_in[0];
  float* out = (float*)d_out;

  // Detect param flattening: sorted-keys (b,beta,gamma,w) vs insertion (w,b,gamma,beta).
  // First layer: w has 32*9 = 288 elems, b/beta/gamma have 32.
  bool w_first = (n_in > 2) && (in_sizes[1] > in_sizes[2]);
  const int IB    = w_first ? 1 : 0;
  const int IBETA = w_first ? 3 : 1;
  const int IGAM  = 2;                 // gamma is 3rd in both orders
  const int IW    = w_first ? 0 : 3;

  // workspace bump allocator
  char* ws = (char*)d_ws;
  size_t off = 0;
  auto alloc = [&](size_t bytes) -> void* {
    void* p = ws + off;
    off += (bytes + 255) & ~(size_t)255;
    return p;
  };

  float* cxyz0 = (float*)alloc((size_t)B * N0 * 3 * 4);
  float* cfeat0 = (float*)alloc((size_t)B * N0 * 6 * 4);
  const int Sst[3] = {512, 128, 64};
  const int Ctot[3] = {320, 512, 512};
  float* nxyz[3]; float* feat[3];
  for (int s = 0; s < 3; ++s) {
    nxyz[s] = (float*)alloc((size_t)B * Sst[s] * 3 * 4);
    feat[s] = (float*)alloc((size_t)B * Sst[s] * Ctot[s] * 4);
  }
  int* fidx = (int*)alloc((size_t)B * 512 * 4);
  int* bidx = (int*)alloc((size_t)B * 512 * 128 * 4);
  __bf16* wbuf = (__bf16*)alloc((size_t)256 * 544 * 2);
  float* stats = (float*)alloc((size_t)2048 * 4);          // sum|sumsq|mean|rstd @ strides of 512
  __bf16* XA = (__bf16*)alloc((size_t)524288 * 128 * 2);   // 134 MB
  __bf16* XB = (__bf16*)alloc((size_t)524288 * 128 * 2);   // 134 MB
  float*  Yb = (float*)alloc((size_t)524288 * 128 * 4);    // 268 MB

  // output offsets (floats), in reference return order
  size_t oo = 0;
  float* out0 = out + oo; oo += (size_t)B * 3 * N0;
  float* out1 = out + oo; oo += (size_t)B * 6 * N0;
  float* outx[3]; float* outf[3];
  for (int s = 0; s < 3; ++s) {
    outx[s] = out + oo; oo += (size_t)B * 3 * Sst[s];
    outf[s] = out + oo; oo += (size_t)B * Ctot[s] * Sst[s];
  }

  // network config
  const int nBr[3] = {3, 2, 2};
  const float radii[3][3] = {{0.1f, 0.2f, 0.4f}, {0.4f, 0.8f, 0.f}, {0.2f, 0.4f, 0.f}};
  const int Ks[3][3] = {{32, 64, 128}, {64, 128, 0}, {16, 32, 0}};
  const int mlp[3][3][3] = {{{32, 32, 64}, {64, 64, 128}, {64, 96, 128}},
                            {{128, 128, 256}, {128, 196, 256}, {0, 0, 0}},
                            {{128, 128, 256}, {128, 196, 256}, {0, 0, 0}}};

  // stage 0 input prep
  k_prep<<<CEILDIV(B * N0, 256), 256, 0, stream>>>(xin, cxyz0, cfeat0, out0, out1, B, N0);

  const float* cur_xyz = cxyz0;
  const float* cur_feat = cfeat0;
  int Nin = N0, Cp = 6;
  int L = 0;   // global layer counter into param list

  for (int st = 0; st < 3; ++st) {
    int S = Sst[st];
    k_fps<<<B, 256, 0, stream>>>(cur_xyz, Nin, S, fidx);
    k_newxyz<<<CEILDIV(B * S, 256), 256, 0, stream>>>(cur_xyz, fidx, nxyz[st], outx[st], B, Nin, S);

    int choff = 0;
    for (int br = 0; br < nBr[st]; ++br) {
      int K = Ks[st][br];
      float r = radii[st][br];
      int M = B * S * K;                     // always a multiple of 16
      k_ball<<<CEILDIV(B * S, 128), 128, 0, stream>>>(cur_xyz, nxyz[st], bidx, B, Nin, S, K, r * r);

      int cin = Cp + 3;
      int CP = cp32(cin);
      k_group<<<CEILDIV(M, 256), 256, 0, stream>>>(cur_xyz, cur_feat, nxyz[st], bidx, XA,
                                                   B, Nin, Cp, S, K, CP);
      __bf16* Xc = XA; __bf16* Xn = XB;

      for (int l = 0; l < 3; ++l) {
        int O = mlp[st][br][l];
        int NPad = np16(O);
        int CPo = cp32(O);
        const float* pb    = (const float*)d_in[1 + 4 * L + IB];
        const float* pbeta = (const float*)d_in[1 + 4 * L + IBETA];
        const float* pgam  = (const float*)d_in[1 + 4 * L + IGAM];
        const float* pw    = (const float*)d_in[1 + 4 * L + IW];
        ++L;

        k_castw<<<CEILDIV(NPad, 64), 64, 0, stream>>>(pw, wbuf, O, cin, CP, NPad);
        launch_gemm(Xc, wbuf, pb, Yb, M, CP, NPad, O, stream);

        hipMemsetAsync(stats, 0, 2 * 512 * sizeof(float), stream);
        dim3 gs(O, CEILDIV(M, 8192));
        k_colstats<<<gs, 256, 0, stream>>>(Yb, M, NPad, stats, stats + 512);
        k_finstats<<<CEILDIV(O, 64), 64, 0, stream>>>(stats, stats + 512, M, O,
                                                      stats + 1024, stats + 1536);
        if (l < 2) {
          k_bnrelu<<<CEILDIV(M, 256), 256, 0, stream>>>(Yb, stats + 1024, stats + 1536,
                                                        pgam, pbeta, Xn, M, NPad, O, CPo);
          __bf16* t = Xc; Xc = Xn; Xn = t;
          cin = O; CP = CPo;
        } else {
          k_maxk<<<CEILDIV(B * S * O, 256), 256, 0, stream>>>(Yb, stats + 1024, stats + 1536,
                                                              pgam, pbeta, feat[st],
                                                              B * S, K, NPad, O, Ctot[st], choff);
          choff += O;
        }
      }
    }
    k_featout<<<CEILDIV(B * S * Ctot[st], 256), 256, 0, stream>>>(feat[st], outf[st], B, S, Ctot[st]);
    cur_xyz = nxyz[st];
    cur_feat = feat[st];
    Nin = S;
    Cp = Ctot[st];
  }
  (void)ws_size; (void)out_size; (void)n_in;
}
